// State_Aware_Dual_Attention_2551210574589
// MI455X (gfx1250) — matrix operand order repaired and checked by reading, not compiled, not run on hardware
//
#include <hip/hip_runtime.h>
#include <hip/hip_bf16.h>

// ---------------------------------------------------------------------------
// Dual attention (softmax + Gaussian positional branch) for MI455X (gfx1250).
// GEMMs and attention use v_wmma_f32_16x16x32_bf16; global->LDS staging goes
// through the Tensor Data Mover where available.
// ---------------------------------------------------------------------------

typedef __attribute__((ext_vector_type(16))) __bf16        v16bf;
typedef __attribute__((ext_vector_type(8)))  __bf16        v8bf;
typedef __attribute__((ext_vector_type(8)))  float         v8f;
typedef __attribute__((ext_vector_type(4)))  unsigned int  u32x4;
typedef __attribute__((ext_vector_type(8)))  int           i32x8;
typedef __attribute__((ext_vector_type(4)))  int           i32x4;

#define B_      4
#define N_      2048
#define DIM_    512
#define HEADS_  8
#define DH_     64
#define INNER_  512            // HEADS_*DH_
#define QKVG_   (4*INNER_)     // 2048
#define CONCAT_ (2*INNER_)     // 1024

#if defined(__HIP_DEVICE_COMPILE__) && \
    __has_builtin(__builtin_amdgcn_tensor_load_to_lds) && \
    __has_builtin(__builtin_amdgcn_s_wait_tensorcnt)
#define USE_TDM 1
#else
#define USE_TDM 0
#endif

__device__ __forceinline__ __bf16 f2bf(float x) { return (__bf16)x; }
__device__ __forceinline__ float  bf2f(__bf16 x){ return (float)x; }

// 32 contiguous bytes -> one fragment (B-fragment / frag-ready layouts).
__device__ __forceinline__ v16bf ld_frag32(const __bf16* p) {
  v8bf lo = *(const v8bf*)p;
  v8bf hi = *(const v8bf*)(p + 8);
  return __builtin_shufflevector(lo, hi, 0,1,2,3,4,5,6,7,8,9,10,11,12,13,14,15);
}
// A-fragment from row-major storage: chunks at +0 and +8 elements.
__device__ __forceinline__ v16bf ld_frag_a(const __bf16* p) {
  v8bf lo = *(const v8bf*)p;
  v8bf hi = *(const v8bf*)(p + 8);
  return __builtin_shufflevector(lo, hi, 0,1,2,3,4,5,6,7,8,9,10,11,12,13,14,15);
}

#if USE_TDM
// Tensor DMA: global (bf16 elements) -> LDS. 1-D (tile_d1==0) or 2-D tile.
// D# per CDNA5 ISA. data_size code 1 == 2 bytes.
// 6-arg builtin variant (clang-23 / therock-10.0): trailing groups zero-filled
// (<=2-D tensor -> VADDR2/3 groups are the NULL configuration).
__device__ __forceinline__ void tdm_load_bf16(
    unsigned lds_off, const void* gaddr,
    unsigned tile_d0, unsigned tile_d1,
    unsigned tensor_d0, unsigned tensor_d1,
    unsigned long long stride0_elems) {
  unsigned long long ga = (unsigned long long)gaddr;
  u32x4 g0;
  g0[0] = 1u;                                             // count=1
  g0[1] = lds_off;                                        // lds_addr (bytes)
  g0[2] = (unsigned)ga;                                   // global_addr[31:0]
  g0[3] = (unsigned)((ga >> 32) & 0x01FFFFFFu) | (2u << 30);  // [56:32]|type=2
  i32x8 g1;
  g1[0] = 0x10000;                                        // data_size = 2B
  g1[1] = (int)((tensor_d0 & 0xFFFFu) << 16);             // tensor_dim0 lo
  g1[2] = (int)(((tensor_d0 >> 16) & 0xFFFFu) |
                ((tensor_d1 & 0xFFFFu) << 16));           // td0 hi | td1 lo
  g1[3] = (int)(((tensor_d1 >> 16) & 0xFFFFu) |
                ((tile_d0 & 0xFFFFu) << 16));             // td1 hi | tile_d0
  g1[4] = (int)(tile_d1 & 0xFFFFu);                       // tile_d1 (tile_d2=0)
  g1[5] = (int)(unsigned)(stride0_elems & 0xFFFFFFFFu);   // stride0 lo
  g1[6] = (int)(unsigned)((stride0_elems >> 32) & 0xFFFFu); // stride0 hi
  g1[7] = 0;
  i32x4 z4; z4[0] = 0; z4[1] = 0; z4[2] = 0; z4[3] = 0;
  i32x8 z8; z8[0] = 0; z8[1] = 0; z8[2] = 0; z8[3] = 0;
  z8[4] = 0; z8[5] = 0; z8[6] = 0; z8[7] = 0;
  __builtin_amdgcn_tensor_load_to_lds(g0, g1, z4, z4, z8, 0);
}
#endif

// ---------------------------------------------------------------------------
// fp32 -> bf16 conversion (x)
// ---------------------------------------------------------------------------
__global__ void cvt_f32_bf16(const float* __restrict__ in,
                             __bf16* __restrict__ out, int n) {
  int i = blockIdx.x * 256 + threadIdx.x;
  if (i < n) out[i] = (__bf16)in[i];
}

// ---------------------------------------------------------------------------
// Pack weight [K, NC] fp32 -> bf16 B-fragment-ready layout:
//   off = (ktile*(NC/16) + ntile)*512 + ((kk&16) + n)*16 + (kk&15)
// One-time cost; makes every B fragment a contiguous 32B run per lane and
// every per-k-step B slab a contiguous 4KB block (TDM/LDS friendly).
// ---------------------------------------------------------------------------
__global__ void pack_w_frag(const float* __restrict__ W, __bf16* __restrict__ Wf,
                            int K, int NC) {
  int tid = blockIdx.x * 256 + threadIdx.x;
  if (tid >= K * NC) return;
  int k = tid / NC, c = tid % NC;
  int kt = k >> 5, kk = k & 31, g = c >> 4, n = c & 15;
  size_t off = ((size_t)(kt * (NC >> 4) + g) << 9) + (size_t)(((kk & 16) + n) << 4) + (kk & 15);
  Wf[off] = (__bf16)W[tid];
}

// ---------------------------------------------------------------------------
// qkvg = x @ w_qkvg. Block: 256 threads (8 waves), block tile 128x64,
// wave tile 16x64. A(128x32, strided) and B(4KB contiguous, frag-ready)
// staged into LDS by the Tensor Data Mover each k-step.
// q/k scattered row-major per head; vg written in V-fragment layout for the
// flash kernel; vs row-major for the Gaussian branch.
// ---------------------------------------------------------------------------
__global__ __launch_bounds__(256)
void gemm_qkvg(const __bf16* __restrict__ X,   // [8192, 512] row-major
               const __bf16* __restrict__ Wf,  // frag-ready [K/32][128][512]
               __bf16* __restrict__ qb, __bf16* __restrict__ kq,
               __bf16* __restrict__ vf, __bf16* __restrict__ vsb) {
  __shared__ __bf16 As[128 * 32];
  __shared__ __bf16 Bs[4 * 512];

  const int tilesN = QKVG_ / 64;              // 32
  int tile = blockIdx.x;
  int m0 = (tile / tilesN) * 128;
  int c0 = (tile % tilesN) * 64;

  int t = threadIdx.x;
  int lane = t & 31, wave = t >> 5;
  int lhalf = (lane < 16) ? 0 : 1;
  int lm = lane & 15;

  v8f acc[4] = {v8f{}, v8f{}, v8f{}, v8f{}};

  for (int k0 = 0; k0 < DIM_; k0 += 32) {
    __syncthreads();
    const __bf16* bsrc = Wf + ((size_t)((k0 >> 5) * (QKVG_ >> 4) + (c0 >> 4)) << 9);
#if USE_TDM
    if (wave == 0) {
      tdm_load_bf16((unsigned)(unsigned long long)As, X + (size_t)m0 * DIM_ + k0,
                    32, 128, 32, 128, DIM_);
      tdm_load_bf16((unsigned)(unsigned long long)Bs, bsrc,
                    2048, 0, 2048, 1, 2048);
      __builtin_amdgcn_s_wait_tensorcnt(0);
    }
#else
    {  // stage A: 128x32
      int idx = t * 16;
      int m = idx >> 5, kk = idx & 31;
      const __bf16* src = X + (size_t)(m0 + m) * DIM_ + k0 + kk;
      ((v8bf*)(As + idx))[0] = ((const v8bf*)src)[0];
      ((v8bf*)(As + idx))[1] = ((const v8bf*)src)[1];
    }
    {  // stage B: contiguous 4KB
      int idx = t * 8;
      *(v8bf*)(Bs + idx) = *(const v8bf*)(bsrc + idx);
    }
#endif
    __syncthreads();

    v16bf a = ld_frag_a(As + (wave * 16 + lm) * 32 + lhalf * 16);
#pragma unroll
    for (int g = 0; g < 4; ++g) {
      v16bf b = ld_frag32(Bs + g * 512 + lane * 16);
      acc[g] = __builtin_amdgcn_wmma_f32_16x16x32_bf16(
          false, a, false, b, (short)0, acc[g], false, false);
    }
  }

  // Epilogue: 64-wide, 64-aligned tile -> single (part, head).
  int part = c0 >> 9;                 // 0:q 1:k 2:vg 3:vs
  int h = (c0 & 511) >> 6;
  if (part == 2) {
    // V-fragment layout per (b,h): off = ((bh*64 + j/32)*4 + g)*512
    //                                   + ((j&16) + n)*16 + (j&15)
#pragma unroll
    for (int g = 0; g < 4; ++g) {
#pragma unroll
      for (int v = 0; v < 8; ++v) {
        int m = wave * 16 + v + lhalf * 8;
        int row = m0 + m;
        int b_i = row >> 11, j = row & (N_ - 1);
        int bh = b_i * HEADS_ + h;
        size_t off = (((size_t)bh * 64 + (j >> 5)) * 4 + g) * 512 +
                     (size_t)(((j & 16) + lm) << 4) + (j & 15);
        vf[off] = f2bf(acc[g][v]);
      }
    }
  } else {
    float scale = (part == 0) ? 0.125f : 1.0f;   // fold dh^-0.5 into q
    __bf16* dst = (part == 0) ? qb : (part == 1) ? kq : vsb;
#pragma unroll
    for (int g = 0; g < 4; ++g) {
#pragma unroll
      for (int v = 0; v < 8; ++v) {
        int m = wave * 16 + v + lhalf * 8;
        int row = m0 + m;
        int b_i = row >> 11, n_i = row & (N_ - 1);
        int d = g * 16 + lm;
        dst[(((size_t)(b_i * HEADS_ + h) * N_) + n_i) * DH_ + d] =
            f2bf(acc[g][v] * scale);
      }
    }
  }
}

// ---------------------------------------------------------------------------
// Flash attention (softmax branch). One wave per 16-row tile; j-steps of 32.
// K fragments and V fragments load straight from global (contiguous 32B/lane,
// L2-resident). LDS only bounces P from C-layout to A-layout.
// ---------------------------------------------------------------------------
__global__ __launch_bounds__(256)
void flash_attn(const __bf16* __restrict__ qb, const __bf16* __restrict__ kq,
                const __bf16* __restrict__ vf, __bf16* __restrict__ att) {
  __shared__ __bf16 ldsP[8][16 * 32];

  int t = threadIdx.x, lane = t & 31, wave = t >> 5;
  int lhalf = (lane < 16) ? 0 : 1;
  int lm = lane & 15;

  int tileIdx = blockIdx.x * 8 + wave;   // 0 .. B*H*(N/16)-1
  int itile = tileIdx & 127;             // N/16 = 128
  int bh = tileIdx >> 7;                 // 0..31
  int i0 = itile * 16;

  const __bf16* Q = qb + (size_t)bh * N_ * DH_;
  const __bf16* K = kq + (size_t)bh * N_ * DH_;
  const __bf16* Vb = vf + (size_t)bh * 64 * 4 * 512;

  // Q fragments (d = WMMA K-dim, two 32-halves); q pre-scaled by dh^-0.5.
  v16bf aq[2];
#pragma unroll
  for (int half = 0; half < 2; ++half)
    aq[half] = ld_frag_a(Q + (size_t)(i0 + lm) * DH_ + half * 32 + lhalf * 16);

  float mrow[8], lrow[8];
#pragma unroll
  for (int v = 0; v < 8; ++v) { mrow[v] = -1e30f; lrow[v] = 0.0f; }
  v8f o[4] = {v8f{}, v8f{}, v8f{}, v8f{}};

  for (int j0 = 0; j0 < N_; j0 += 32) {
    // S tiles: S[m,n] = sum_d q[i0+m,d] * k[j0+16t+n,d]
    v8f s[2];
#pragma unroll
    for (int tt = 0; tt < 2; ++tt) {
      v8f c = {};
#pragma unroll
      for (int half = 0; half < 2; ++half) {
        v16bf bk = ld_frag32(
            K + (size_t)(j0 + tt * 16 + lm) * DH_ + half * 32 + lhalf * 16);
        c = __builtin_amdgcn_wmma_f32_16x16x32_bf16(
            false, aq[half], false, bk, (short)0, c, false, false);
      }
      s[tt] = c;
    }

    // Online softmax. Row m = v + lhalf*8 lives in 16 lanes of one half.
    float p0[8], p1[8];
#pragma unroll
    for (int v = 0; v < 8; ++v) {
      float mx = fmaxf(s[0][v], s[1][v]);
#pragma unroll
      for (int off = 1; off < 16; off <<= 1)
        mx = fmaxf(mx, __shfl_xor(mx, off, 32));
      float mnew = fmaxf(mrow[v], mx);
      float alpha = __expf(mrow[v] - mnew);
      float a0 = __expf(s[0][v] - mnew);
      float a1 = __expf(s[1][v] - mnew);
      float rs = a0 + a1;
#pragma unroll
      for (int off = 1; off < 16; off <<= 1)
        rs += __shfl_xor(rs, off, 32);
      lrow[v] = lrow[v] * alpha + rs;
      mrow[v] = mnew;
      p0[v] = a0; p1[v] = a1;
#pragma unroll
      for (int g = 0; g < 4; ++g) o[g][v] *= alpha;
    }

    // P (C-layout) -> LDS [m][kk]; per-wave private region.
#pragma unroll
    for (int v = 0; v < 8; ++v) {
      int m = v + lhalf * 8;
      ldsP[wave][m * 32 + lm]      = f2bf(p0[v]);
      ldsP[wave][m * 32 + 16 + lm] = f2bf(p1[v]);
    }
    __syncthreads();
    v16bf ap = ld_frag_a(&ldsP[wave][lm * 32 + lhalf * 16]);

    // O += P @ V : V fragments direct from global (fragment-ready layout).
#pragma unroll
    for (int g = 0; g < 4; ++g) {
      v16bf bv = ld_frag32(Vb + (((size_t)(j0 >> 5)) * 4 + g) * 512 + lane * 16);
      o[g] = __builtin_amdgcn_wmma_f32_16x16x32_bf16(
          false, ap, false, bv, (short)0, o[g], false, false);
    }
    __syncthreads();
  }

  // Epilogue: out_g columns [h*64 .. h*64+64) of concat buffer [B*N, 1024].
  int b_i = bh >> 3, h = bh & 7;
#pragma unroll
  for (int g = 0; g < 4; ++g) {
#pragma unroll
    for (int v = 0; v < 8; ++v) {
      int m = v + lhalf * 8;
      size_t row = (size_t)b_i * N_ + i0 + m;
      int col = h * 64 + g * 16 + lm;
      att[row * CONCAT_ + col] = f2bf(o[g][v] / lrow[v]);
    }
  }
}

// ---------------------------------------------------------------------------
// Gaussian positional branch: normalized 65-tap windowed convolution.
// exp(-d^2/(2*1.92^2)) < 1e-15 for |d| >= 16 -> window is exact to fp32.
// The 1/(sqrt(2pi)*sigma) factor cancels in the normalization.
// ---------------------------------------------------------------------------
__global__ __launch_bounds__(256)
void gauss_attn(const __bf16* __restrict__ vsb, const float* __restrict__ sigma,
                __bf16* __restrict__ att) {
  int t = threadIdx.x;
  int d = t & 63;
  int ridx = blockIdx.x * 4 + (t >> 6);   // (b,h,i) flattened
  int i = ridx & (N_ - 1);
  int bh = ridx >> 11;
  int h = bh & 7;

  float sg = sigma[h];
  float inv2s2 = 1.0f / (2.0f * sg * sg);

  int jlo = i - 32; if (jlo < 0) jlo = 0;
  int jhi = i + 32; if (jhi > N_ - 1) jhi = N_ - 1;

  const __bf16* vrow = vsb + ((size_t)bh * N_) * DH_ + d;
  float num = 0.0f, den = 0.0f;
  for (int j = jlo; j <= jhi; ++j) {
    float diff = (float)(i - j);
    float w = __expf(-diff * diff * inv2s2);
    num += w * bf2f(vrow[(size_t)j * DH_]);
    den += w;
  }
  int b_i = bh >> 3;
  att[((size_t)b_i * N_ + i) * CONCAT_ + INNER_ + h * 64 + d] = f2bf(num / den);
}

// ---------------------------------------------------------------------------
// out = att[8192,1024] @ w_out[1024,512] + b_out  (fp32 result)
// ---------------------------------------------------------------------------
__global__ __launch_bounds__(256)
void gemm_out(const __bf16* __restrict__ A_, const __bf16* __restrict__ Wf,
              const float* __restrict__ bias, float* __restrict__ out) {
  __shared__ __bf16 As[128 * 32];
  __shared__ __bf16 Bs[4 * 512];

  const int KDIM = CONCAT_;      // 1024
  const int NCOL = DIM_;         // 512
  const int tilesN = NCOL / 64;  // 8
  int tile = blockIdx.x;
  int m0 = (tile / tilesN) * 128;
  int c0 = (tile % tilesN) * 64;

  int t = threadIdx.x;
  int lane = t & 31, wave = t >> 5;
  int lhalf = (lane < 16) ? 0 : 1;
  int lm = lane & 15;

  v8f acc[4] = {v8f{}, v8f{}, v8f{}, v8f{}};

  for (int k0 = 0; k0 < KDIM; k0 += 32) {
    __syncthreads();
    const __bf16* bsrc = Wf + ((size_t)((k0 >> 5) * (NCOL >> 4) + (c0 >> 4)) << 9);
#if USE_TDM
    if (wave == 0) {
      tdm_load_bf16((unsigned)(unsigned long long)As, A_ + (size_t)m0 * KDIM + k0,
                    32, 128, 32, 128, KDIM);
      tdm_load_bf16((unsigned)(unsigned long long)Bs, bsrc,
                    2048, 0, 2048, 1, 2048);
      __builtin_amdgcn_s_wait_tensorcnt(0);
    }
#else
    {
      int idx = t * 16;
      int m = idx >> 5, kk = idx & 31;
      const __bf16* src = A_ + (size_t)(m0 + m) * KDIM + k0 + kk;
      ((v8bf*)(As + idx))[0] = ((const v8bf*)src)[0];
      ((v8bf*)(As + idx))[1] = ((const v8bf*)src)[1];
    }
    {
      int idx = t * 8;
      *(v8bf*)(Bs + idx) = *(const v8bf*)(bsrc + idx);
    }
#endif
    __syncthreads();

    v16bf a = ld_frag_a(As + (wave * 16 + lm) * 32 + lhalf * 16);
#pragma unroll
    for (int g = 0; g < 4; ++g) {
      v16bf b = ld_frag32(Bs + g * 512 + lane * 16);
      acc[g] = __builtin_amdgcn_wmma_f32_16x16x32_bf16(
          false, a, false, b, (short)0, acc[g], false, false);
    }
  }

#pragma unroll
  for (int g = 0; g < 4; ++g) {
#pragma unroll
    for (int v = 0; v < 8; ++v) {
      int m = wave * 16 + v + lhalf * 8;
      size_t row = (size_t)(m0 + m);
      int c = c0 + g * 16 + lm;
      out[row * NCOL + c] = acc[g][v] + bias[c];
    }
  }
}

// ---------------------------------------------------------------------------
// Launch
// ---------------------------------------------------------------------------
extern "C" void kernel_launch(void* const* d_in, const int* in_sizes, int n_in,
                              void* d_out, int out_size, void* d_ws, size_t ws_size,
                              hipStream_t stream) {
  const float* x      = (const float*)d_in[0];
  const float* w_qkvg = (const float*)d_in[1];
  const float* sigma  = (const float*)d_in[2];
  const float* w_out  = (const float*)d_in[3];
  const float* b_out  = (const float*)d_in[4];
  float* out = (float*)d_out;

  char* ws = (char*)d_ws;
  size_t off = 0;
  auto carve = [&](size_t bytes) -> void* {
    void* p = ws + off;
    off += (bytes + 255) & ~(size_t)255;
    return p;
  };
  __bf16* xb  = (__bf16*)carve((size_t)B_ * N_ * DIM_ * 2);
  __bf16* wqf = (__bf16*)carve((size_t)DIM_ * QKVG_ * 2);
  __bf16* wof = (__bf16*)carve((size_t)CONCAT_ * DIM_ * 2);
  __bf16* qb  = (__bf16*)carve((size_t)B_ * HEADS_ * N_ * DH_ * 2);
  __bf16* kq  = (__bf16*)carve((size_t)B_ * HEADS_ * N_ * DH_ * 2);
  __bf16* vfb = (__bf16*)carve((size_t)B_ * HEADS_ * N_ * DH_ * 2);
  __bf16* vsb = (__bf16*)carve((size_t)B_ * HEADS_ * N_ * DH_ * 2);
  __bf16* att = (__bf16*)carve((size_t)B_ * N_ * CONCAT_ * 2);

  int nx = B_ * N_ * DIM_;
  cvt_f32_bf16<<<(nx + 255) / 256, 256, 0, stream>>>(x, xb, nx);
  pack_w_frag<<<(DIM_ * QKVG_ + 255) / 256, 256, 0, stream>>>(w_qkvg, wqf, DIM_, QKVG_);
  pack_w_frag<<<(CONCAT_ * DIM_ + 255) / 256, 256, 0, stream>>>(w_out, wof, CONCAT_, DIM_);

  gemm_qkvg<<<2048, 256, 0, stream>>>(xb, wqf, qb, kq, vfb, vsb);
  flash_attn<<<512, 256, 0, stream>>>(qb, kq, vfb, att);
  gauss_attn<<<B_ * HEADS_ * N_ / 4, 256, 0, stream>>>(vsb, sigma, att);
  gemm_out<<<512, 256, 0, stream>>>(att, wof, b_out, out);
}
